// MoEHead_41987600285958
// MI455X (gfx1250) — matrix operand order repaired and checked by reading, not compiled, not run on hardware
//
#include <hip/hip_runtime.h>
#include <hip/hip_bf16.h>
#include <math.h>

// ---------------------------------------------------------------------------
// MoE (top-2 of 8 experts) + residual + LayerNorm for MI455X (gfx1250).
// B=2, T=2048 -> N=4096 tokens, H=1024, F=4096, E=8.
// bf16 WMMA (v_wmma_f32_16x16x32_bf16) GEMMs over per-expert gathered tokens.
// Global->LDS staging uses CDNA5 async copies (global_load_async_to_lds_b128,
// ASYNCcnt) with double-buffered LDS tiles, K-step 64.
// ---------------------------------------------------------------------------

#define NTOK 4096
#define HDIM 1024
#define FDIM 4096
#define NEXP 8
#define TOPK 2
#define LDT  72        // LDS row stride (elements) for K=64 tiles (bank-conflict pad)
#define LN_EPS 1e-5f

typedef __bf16 bf16;
typedef __attribute__((ext_vector_type(4)))  bf16  v4bf;
typedef __attribute__((ext_vector_type(8)))  bf16  v8bf;
typedef __attribute__((ext_vector_type(16))) bf16  v16bf;
typedef __attribute__((ext_vector_type(8)))  float v8f;

// ---- WMMA fragment loads from LDS (layouts per CDNA5 ISA 7.12.2) ----------
// A (16x32 bf16): lane holds row m=lane&15; K chunks [kh..kh+7] and [kh+16..kh+23],
// kh = (lane>=16)?8:0.
static __device__ inline v16bf frag_a(const bf16* As, int mbase, int lane) {
  const bf16* p = As + (mbase + (lane & 15)) * LDT + ((lane >> 4) << 3);
  v8bf lo = *(const v8bf*)p;
  v8bf hi = *(const v8bf*)(p + 16);
  return __builtin_shufflevector(lo, hi, 0,1,2,3,4,5,6,7,8,9,10,11,12,13,14,15);
}
// B (32x16 bf16), stored N-major in LDS (Bs[n][k]): lane holds col n=lane&15;
// K chunks [kh..kh+7] and [kh+16..kh+23], kh = (lane>=16)?8:0.
static __device__ inline v16bf frag_b(const bf16* Bs, int nbase, int lane) {
  const bf16* p = Bs + (nbase + (lane & 15)) * LDT + ((lane >> 4) << 3);
  v8bf lo = *(const v8bf*)p;
  v8bf hi = *(const v8bf*)(p + 16);
  return __builtin_shufflevector(lo, hi, 0,1,2,3,4,5,6,7,8,9,10,11,12,13,14,15);
}

static __device__ inline v8f wmma_bf16(v16bf a, v16bf b, v8f c) {
  return __builtin_amdgcn_wmma_f32_16x16x32_bf16(false, a, false, b, (short)0, c,
                                                 false, false);
}

// ---- CDNA5 async global->LDS copy: 64 contiguous bytes per thread ---------
// LDS and global addresses are each a VGPR base plus the immediate offset
// (same offset applies to both sides -> 4x b128 with offset:0/16/32/48).
static __device__ inline void async_copy64(unsigned lds, unsigned long long g) {
  asm volatile(
      "global_load_async_to_lds_b128 %0, %1, off\n\t"
      "global_load_async_to_lds_b128 %0, %1, off offset:16\n\t"
      "global_load_async_to_lds_b128 %0, %1, off offset:32\n\t"
      "global_load_async_to_lds_b128 %0, %1, off offset:48"
      :: "v"(lds), "v"(g) : "memory");
}
#define WAIT_ASYNC_8() asm volatile("s_wait_asynccnt 0x8" ::: "memory")
#define WAIT_ASYNC_0() asm volatile("s_wait_asynccnt 0x0" ::: "memory")

static __device__ inline unsigned lds_off(const void* p) {
  return (unsigned)(size_t)p;   // low 32 bits of a generic LDS pointer = LDS byte addr
}

// ---------------------------------------------------------------------------
// Router: logits = x @ rw^T + rb ; top-2 softmax; atomic-append to expert
// lists; also converts x -> bf16 (xb). One block per token, 256 threads.
// ---------------------------------------------------------------------------
__global__ __launch_bounds__(256) void moe_router_kernel(
    const float* __restrict__ x, const float* __restrict__ rw,
    const float* __restrict__ rb, bf16* __restrict__ xb,
    int* __restrict__ cnt, int* __restrict__ tokl, float* __restrict__ gatel)
{
  const int t = blockIdx.x;
  const int tid = threadIdx.x;
  const int lane = tid & 31, wave = tid >> 5;

  const float4 xv = ((const float4*)(x + (size_t)t * HDIM))[tid];
  v4bf xo = { (bf16)xv.x, (bf16)xv.y, (bf16)xv.z, (bf16)xv.w };
  *(v4bf*)(xb + (size_t)t * HDIM + tid * 4) = xo;

  float part[NEXP];
#pragma unroll
  for (int e = 0; e < NEXP; ++e) {
    const float4 wv = ((const float4*)(rw + (size_t)e * HDIM))[tid];
    part[e] = fmaf(xv.x, wv.x, fmaf(xv.y, wv.y, fmaf(xv.z, wv.z, xv.w * wv.w)));
  }
#pragma unroll
  for (int off = 16; off >= 1; off >>= 1)
#pragma unroll
    for (int e = 0; e < NEXP; ++e) part[e] += __shfl_xor(part[e], off, 32);

  __shared__ float red[8][NEXP];
  if (lane == 0)
#pragma unroll
    for (int e = 0; e < NEXP; ++e) red[wave][e] = part[e];
  __syncthreads();

  if (tid == 0) {
    float logit[NEXP];
#pragma unroll
    for (int e = 0; e < NEXP; ++e) {
      float s = rb[e];
#pragma unroll
      for (int w = 0; w < 8; ++w) s += red[w][e];
      logit[e] = s;
    }
    int i0 = 0;
#pragma unroll
    for (int e = 1; e < NEXP; ++e) if (logit[e] > logit[i0]) i0 = e;
    int i1 = (i0 == 0) ? 1 : 0;
#pragma unroll
    for (int e = 0; e < NEXP; ++e)
      if (e != i0 && logit[e] > logit[i1]) i1 = e;
    const float m = logit[i0];
    const float e0 = expf(logit[i0] - m), e1 = expf(logit[i1] - m);
    const float inv = 1.0f / (e0 + e1);
    int p0 = atomicAdd(&cnt[i0], 1);
    tokl[i0 * NTOK + p0] = t;                 // slot 0
    gatel[i0 * NTOK + p0] = e0 * inv;
    int p1 = atomicAdd(&cnt[i1], 1);
    tokl[i1 * NTOK + p1] = t | (1 << 16);     // slot 1
    gatel[i1 * NTOK + p1] = e1 * inv;
  }
}

__global__ void moe_scan_kernel(const int* __restrict__ cnt, int* __restrict__ hoff) {
  if (threadIdx.x == 0 && blockIdx.x == 0) {
    int s = 0;
    for (int e = 0; e < NEXP; ++e) { hoff[e] = s; s += cnt[e]; }
  }
}

// ---------------------------------------------------------------------------
// fp32 [R,C] -> bf16 [C,R] (batched over blockIdx.z). 32x32 LDS tile.
// Produces N-major weights so WMMA B fragments are contiguous in LDS/global.
// ---------------------------------------------------------------------------
__global__ __launch_bounds__(256) void moe_cvt_transpose_kernel(
    const float* __restrict__ src, bf16* __restrict__ dst, int R, int C)
{
  __shared__ float tile[32][33];
  const size_t base = (size_t)blockIdx.z * R * C;
  const int c0 = blockIdx.x * 32, r0 = blockIdx.y * 32;
  const int tx = threadIdx.x & 31, ty = threadIdx.x >> 5;
#pragma unroll
  for (int i = 0; i < 4; ++i)
    tile[ty + i * 8][tx] = src[base + (size_t)(r0 + ty + i * 8) * C + c0 + tx];
  __syncthreads();
#pragma unroll
  for (int i = 0; i < 4; ++i)
    dst[base + (size_t)(c0 + ty + i * 8) * R + r0 + tx] = (bf16)tile[tx][ty + i * 8];
}

// ---------------------------------------------------------------------------
// GEMM1: h[hoff[e]+m, :] = gelu( xb[tok[e][m], :] @ W1t[e]^T + b1[e] ), bf16 out.
// 128x128 tile, 8 waves, each wave 64x32 (4x2 WMMA frags).
// Double-buffered K-step 64 staged with async global->LDS copies.
// ---------------------------------------------------------------------------
__global__ __launch_bounds__(256) void moe_gemm1_kernel(
    const bf16* __restrict__ xb, const bf16* __restrict__ w1t,
    const float* __restrict__ b1, const int* __restrict__ cnt,
    const int* __restrict__ hoff, const int* __restrict__ tokl,
    bf16* __restrict__ hbuf)
{
  const int e = blockIdx.z;
  const int M = cnt[e];
  const int row0 = blockIdx.y * 128;
  if (row0 >= M) return;
  const int col0 = blockIdx.x * 128;
  const int hbase = hoff[e];

  __shared__ __align__(16) bf16 As[2][128 * LDT];
  __shared__ __align__(16) bf16 Bs[2][128 * LDT];
  __shared__ int toks[128];

  const int tid = threadIdx.x;
  const int lane = tid & 31;
  const int wave = tid >> 5;
  const int wm = (wave >> 2) * 64;   // wave row base in tile
  const int wn = (wave & 3) * 32;    // wave col base in tile

  if (tid < 128) {
    int m = row0 + tid;
    if (m >= M) m = M - 1;           // clamp (compute garbage, masked on store)
    toks[tid] = tokl[e * NTOK + m];
  }
  __syncthreads();

  v8f acc[4][2] = {};

  const int r    = tid >> 1;         // tile row this thread stages
  const int koff = (tid & 1) << 5;   // element offset 0 / 32 within K=64
  const unsigned long long gA =
      (unsigned long long)(xb + (size_t)(toks[r] & 0xFFFF) * HDIM + koff);
  const unsigned long long gB =
      (unsigned long long)(w1t + ((size_t)e * FDIM + col0 + r) * HDIM + koff);
  unsigned lA[2], lB[2];
  lA[0] = lds_off(&As[0][r * LDT + koff]);
  lA[1] = lds_off(&As[1][r * LDT + koff]);
  lB[0] = lds_off(&Bs[0][r * LDT + koff]);
  lB[1] = lds_off(&Bs[1][r * LDT + koff]);

  async_copy64(lA[0], gA);           // stage 0 (8 async b128 / thread incl. B)
  async_copy64(lB[0], gB);

  const int NSTAGE = HDIM / 64;
  for (int s = 0; s < NSTAGE; ++s) {
    const int buf = s & 1;
    if (s + 1 < NSTAGE) {
      const size_t adv = (size_t)(s + 1) * 128;   // 64 elts * 2B per stage
      async_copy64(lA[buf ^ 1], gA + adv);
      async_copy64(lB[buf ^ 1], gB + adv);
      WAIT_ASYNC_8();                // current stage's 8 ops complete (in-order)
    } else {
      WAIT_ASYNC_0();
    }
    __syncthreads();                 // stage data visible to all waves

    const bf16* Ab = &As[buf][0];
    const bf16* Bb = &Bs[buf][0];
#pragma unroll
    for (int kk = 0; kk < 64; kk += 32) {
      v16bf a0 = frag_a(Ab + kk, wm + 0,  lane);
      v16bf a1 = frag_a(Ab + kk, wm + 16, lane);
      v16bf a2 = frag_a(Ab + kk, wm + 32, lane);
      v16bf a3 = frag_a(Ab + kk, wm + 48, lane);
      v16bf bb0 = frag_b(Bb + kk, wn + 0,  lane);
      v16bf bb1 = frag_b(Bb + kk, wn + 16, lane);
      acc[0][0] = wmma_bf16(a0, bb0, acc[0][0]);
      acc[0][1] = wmma_bf16(a0, bb1, acc[0][1]);
      acc[1][0] = wmma_bf16(a1, bb0, acc[1][0]);
      acc[1][1] = wmma_bf16(a1, bb1, acc[1][1]);
      acc[2][0] = wmma_bf16(a2, bb0, acc[2][0]);
      acc[2][1] = wmma_bf16(a2, bb1, acc[2][1]);
      acc[3][0] = wmma_bf16(a3, bb0, acc[3][0]);
      acc[3][1] = wmma_bf16(a3, bb1, acc[3][1]);
    }
    __syncthreads();                 // all waves done reading buf
  }

  const int khi = (lane >> 4) << 3;  // C/D layout: lanes>=16 hold rows r+8
#pragma unroll
  for (int mi = 0; mi < 4; ++mi) {
    const int lrb = wm + mi * 16 + khi;
#pragma unroll
    for (int ni = 0; ni < 2; ++ni) {
      const int col = col0 + wn + ni * 16 + (lane & 15);
      const float bias = b1[e * FDIM + col];
      v8f a = acc[mi][ni];
#pragma unroll
      for (int rr = 0; rr < 8; ++rr) {
        const int lrow = lrb + rr;
        if (row0 + lrow < M) {
          float v = a[rr] + bias;
          float g = 0.5f * v * (1.0f + erff(v * 0.70710678118f));  // exact gelu
          hbuf[(size_t)(hbase + row0 + lrow) * FDIM + col] = (bf16)g;
        }
      }
    }
  }
}

// ---------------------------------------------------------------------------
// GEMM2: y = h_e @ W2t[e]^T + b2[e];  out_partial[slot][token,:] = gate * y.
// Each (token,slot) written exactly once -> no atomics / no init needed.
// ---------------------------------------------------------------------------
__global__ __launch_bounds__(256) void moe_gemm2_kernel(
    const bf16* __restrict__ hbuf, const bf16* __restrict__ w2t,
    const float* __restrict__ b2, const int* __restrict__ cnt,
    const int* __restrict__ hoff, const int* __restrict__ tokl,
    const float* __restrict__ gatel, float* __restrict__ outp)
{
  const int e = blockIdx.z;
  const int M = cnt[e];
  const int row0 = blockIdx.y * 128;
  if (row0 >= M) return;
  const int col0 = blockIdx.x * 128;
  const int hbase = hoff[e];

  __shared__ __align__(16) bf16 As[2][128 * LDT];
  __shared__ __align__(16) bf16 Bs[2][128 * LDT];
  __shared__ int toks[128];
  __shared__ float gates[128];

  const int tid = threadIdx.x;
  const int lane = tid & 31;
  const int wave = tid >> 5;
  const int wm = (wave >> 2) * 64;
  const int wn = (wave & 3) * 32;

  if (tid < 128) {
    int m = row0 + tid;
    if (m >= M) m = M - 1;
    toks[tid]  = tokl[e * NTOK + m];
    gates[tid] = gatel[e * NTOK + m];
  }
  __syncthreads();

  v8f acc[4][2] = {};

  const int r    = tid >> 1;
  const int koff = (tid & 1) << 5;
  int arow = row0 + r; if (arow >= M) arow = M - 1;
  const unsigned long long gA =
      (unsigned long long)(hbuf + (size_t)(hbase + arow) * FDIM + koff);
  const unsigned long long gB =
      (unsigned long long)(w2t + ((size_t)e * HDIM + col0 + r) * FDIM + koff);
  unsigned lA[2], lB[2];
  lA[0] = lds_off(&As[0][r * LDT + koff]);
  lA[1] = lds_off(&As[1][r * LDT + koff]);
  lB[0] = lds_off(&Bs[0][r * LDT + koff]);
  lB[1] = lds_off(&Bs[1][r * LDT + koff]);

  async_copy64(lA[0], gA);
  async_copy64(lB[0], gB);

  const int NSTAGE = FDIM / 64;
  for (int s = 0; s < NSTAGE; ++s) {
    const int buf = s & 1;
    if (s + 1 < NSTAGE) {
      const size_t adv = (size_t)(s + 1) * 128;
      async_copy64(lA[buf ^ 1], gA + adv);
      async_copy64(lB[buf ^ 1], gB + adv);
      WAIT_ASYNC_8();
    } else {
      WAIT_ASYNC_0();
    }
    __syncthreads();

    const bf16* Ab = &As[buf][0];
    const bf16* Bb = &Bs[buf][0];
#pragma unroll
    for (int kk = 0; kk < 64; kk += 32) {
      v16bf a0 = frag_a(Ab + kk, wm + 0,  lane);
      v16bf a1 = frag_a(Ab + kk, wm + 16, lane);
      v16bf a2 = frag_a(Ab + kk, wm + 32, lane);
      v16bf a3 = frag_a(Ab + kk, wm + 48, lane);
      v16bf bb0 = frag_b(Bb + kk, wn + 0,  lane);
      v16bf bb1 = frag_b(Bb + kk, wn + 16, lane);
      acc[0][0] = wmma_bf16(a0, bb0, acc[0][0]);
      acc[0][1] = wmma_bf16(a0, bb1, acc[0][1]);
      acc[1][0] = wmma_bf16(a1, bb0, acc[1][0]);
      acc[1][1] = wmma_bf16(a1, bb1, acc[1][1]);
      acc[2][0] = wmma_bf16(a2, bb0, acc[2][0]);
      acc[2][1] = wmma_bf16(a2, bb1, acc[2][1]);
      acc[3][0] = wmma_bf16(a3, bb0, acc[3][0]);
      acc[3][1] = wmma_bf16(a3, bb1, acc[3][1]);
    }
    __syncthreads();
  }

  const int khi = (lane >> 4) << 3;
#pragma unroll
  for (int mi = 0; mi < 4; ++mi) {
    const int lrb = wm + mi * 16 + khi;
#pragma unroll
    for (int ni = 0; ni < 2; ++ni) {
      const int col = col0 + wn + ni * 16 + (lane & 15);
      const float bias = b2[e * HDIM + col];
      v8f a = acc[mi][ni];
#pragma unroll
      for (int rr = 0; rr < 8; ++rr) {
        const int lrow = lrb + rr;
        if (row0 + lrow < M) {
          const int te = toks[lrow];
          const int token = te & 0xFFFF;
          const int slot = (te >> 16) & 1;
          outp[((size_t)slot * NTOK + token) * HDIM + col] =
              (a[rr] + bias) * gates[lrow];
        }
      }
    }
  }
}

// ---------------------------------------------------------------------------
// LayerNorm( x + out_partial[0] + out_partial[1] ). One block per token.
// ---------------------------------------------------------------------------
__global__ __launch_bounds__(256) void moe_ln_kernel(
    const float* __restrict__ x, const float* __restrict__ outp,
    const float* __restrict__ gamma, const float* __restrict__ beta,
    float* __restrict__ out)
{
  const int t = blockIdx.x;
  const int tid = threadIdx.x;
  const int lane = tid & 31, wave = tid >> 5;

  const float4 xv = ((const float4*)(x + (size_t)t * HDIM))[tid];
  const float4 p0 = ((const float4*)(outp + (size_t)t * HDIM))[tid];
  const float4 p1 = ((const float4*)(outp + (size_t)(NTOK + t) * HDIM))[tid];
  float v[4] = { xv.x + p0.x + p1.x, xv.y + p0.y + p1.y,
                 xv.z + p0.z + p1.z, xv.w + p0.w + p1.w };
  float s = 0.f, s2 = 0.f;
#pragma unroll
  for (int i = 0; i < 4; ++i) { s += v[i]; s2 += v[i] * v[i]; }
#pragma unroll
  for (int off = 16; off >= 1; off >>= 1) {
    s  += __shfl_xor(s, off, 32);
    s2 += __shfl_xor(s2, off, 32);
  }
  __shared__ float rs[8], rs2[8], stats[2];
  if (lane == 0) { rs[wave] = s; rs2[wave] = s2; }
  __syncthreads();
  if (tid == 0) {
    float S = 0.f, S2 = 0.f;
#pragma unroll
    for (int w = 0; w < 8; ++w) { S += rs[w]; S2 += rs2[w]; }
    const float mu = S / (float)HDIM;
    const float var = S2 / (float)HDIM - mu * mu;
    stats[0] = mu;
    stats[1] = rsqrtf(var + LN_EPS);
  }
  __syncthreads();
  const float mu = stats[0], rstd = stats[1];
  float4 o;
  const int c = tid * 4;
  o.x = (v[0] - mu) * rstd * gamma[c + 0] + beta[c + 0];
  o.y = (v[1] - mu) * rstd * gamma[c + 1] + beta[c + 1];
  o.z = (v[2] - mu) * rstd * gamma[c + 2] + beta[c + 2];
  o.w = (v[3] - mu) * rstd * gamma[c + 3] + beta[c + 3];
  ((float4*)(out + (size_t)t * HDIM))[tid] = o;
}

// ---------------------------------------------------------------------------
extern "C" void kernel_launch(void* const* d_in, const int* in_sizes, int n_in,
                              void* d_out, int out_size, void* d_ws, size_t ws_size,
                              hipStream_t stream)
{
  const float* x   = (const float*)d_in[0];
  const float* rw  = (const float*)d_in[1];
  const float* rb  = (const float*)d_in[2];
  const float* W1  = (const float*)d_in[3];
  const float* b1  = (const float*)d_in[4];
  const float* W2  = (const float*)d_in[5];
  const float* b2  = (const float*)d_in[6];
  const float* gam = (const float*)d_in[7];
  const float* bet = (const float*)d_in[8];
  float* out = (float*)d_out;

  char* p = (char*)d_ws;
  auto take = [&](size_t bytes) {
    char* q = p;
    p += (bytes + 255) & ~(size_t)255;
    return q;
  };
  bf16*  xb    = (bf16*)take((size_t)NTOK * HDIM * sizeof(bf16));
  bf16*  w1t   = (bf16*)take((size_t)NEXP * HDIM * FDIM * sizeof(bf16)); // [e][F][H]
  bf16*  w2t   = (bf16*)take((size_t)NEXP * HDIM * FDIM * sizeof(bf16)); // [e][H][F]
  bf16*  hbuf  = (bf16*)take((size_t)NTOK * TOPK * FDIM * sizeof(bf16));
  float* outp  = (float*)take((size_t)TOPK * NTOK * HDIM * sizeof(float));
  int*   tokl  = (int*)take((size_t)NEXP * NTOK * sizeof(int));
  float* gatel = (float*)take((size_t)NEXP * NTOK * sizeof(float));
  int*   cnt   = (int*)take(256);
  int*   hoff  = (int*)take(256);

  hipMemsetAsync(cnt, 0, NEXP * sizeof(int), stream);
  moe_router_kernel<<<NTOK, 256, 0, stream>>>(x, rw, rb, xb, cnt, tokl, gatel);
  moe_scan_kernel<<<1, 32, 0, stream>>>(cnt, hoff);
  moe_cvt_transpose_kernel<<<dim3(FDIM / 32, HDIM / 32, NEXP), 256, 0, stream>>>(
      W1, w1t, HDIM, FDIM);
  moe_cvt_transpose_kernel<<<dim3(HDIM / 32, FDIM / 32, NEXP), 256, 0, stream>>>(
      W2, w2t, FDIM, HDIM);
  moe_gemm1_kernel<<<dim3(FDIM / 128, NTOK / 128, NEXP), 256, 0, stream>>>(
      xb, w1t, b1, cnt, hoff, tokl, hbuf);
  moe_gemm2_kernel<<<dim3(HDIM / 128, NTOK / 128, NEXP), 256, 0, stream>>>(
      hbuf, w2t, b2, cnt, hoff, tokl, gatel, outp);
  moe_ln_kernel<<<NTOK, 256, 0, stream>>>(x, outp, gam, bet, out);
}
